// VectorQuantizer_62148176773304
// MI455X (gfx1250) — matrix operand order repaired and checked by reading, not compiled, not run on hardware
//
#include <hip/hip_runtime.h>
#include <hip/hip_bf16.h>

typedef __bf16 bf16;
typedef __attribute__((ext_vector_type(8)))  __bf16 v8bf;
typedef __attribute__((ext_vector_type(16))) __bf16 v16bf;
typedef __attribute__((ext_vector_type(8)))  float  v8f;

#define N_TOK   16384
#define DIM     256
#define KCODES  8192
#define BATCH   16
#define HW      1024
#define NUMEL   (N_TOK*DIM)        // 4194304 (x_q elements)
#define BETA_C  0.25f
#define ORTHO_C 10.0f

#define KSPLIT  4
#define CODES_PER_SPLIT (KCODES / KSPLIT)   // 2048
#define ROWS_PER_WG     256                 // 8 waves * 32 rows
#define ROW_BLOCKS      (N_TOK / ROWS_PER_WG)   // 64

// ---- workspace layout (bytes) ----
#define OFF_XHI   (size_t)(0)                        // N*D bf16  (8 MiB)
#define OFF_XLO   (OFF_XHI + (size_t)N_TOK*DIM*2)    // N*D bf16
#define OFF_EHI   (OFF_XLO + (size_t)N_TOK*DIM*2)    // K*D bf16  (4 MiB)
#define OFF_ELO   (OFF_EHI + (size_t)KCODES*DIM*2)   // K*D bf16
#define OFF_HEE   (OFF_ELO + (size_t)KCODES*DIM*2)   // K f32 : 0.5*||e||^2
#define OFF_W2    (OFF_HEE + (size_t)KCODES*4)       // K f32 : 1/||e||^2
#define OFF_IDX   (OFF_W2  + (size_t)KCODES*4)       // N i32 : merged argmin
#define OFF_PVAL  (OFF_IDX + (size_t)N_TOK*4)        // KSPLIT*N f32 candidates
#define OFF_PIDX  (OFF_PVAL + (size_t)KSPLIT*N_TOK*4)// KSPLIT*N i32 candidates
#define OFF_ACC   (OFF_PIDX + (size_t)KSPLIT*N_TOK*4)// f32 accum [0]=mse [1]=||G||^2

// ---------------------------------------------------------------- utilities
__device__ __forceinline__ float block_reduce_sum(float v) {
    #pragma unroll
    for (int o = 16; o > 0; o >>= 1) v += __shfl_xor(v, o, 32);
    __shared__ float red[8];
    if ((threadIdx.x & 31) == 0) red[threadIdx.x >> 5] = v;
    __syncthreads();
    float t = 0.f;
    if (threadIdx.x == 0) {
        #pragma unroll
        for (int i = 0; i < 8; ++i) t += red[i];
    }
    __syncthreads();
    return t;   // valid in thread 0 only
}

#define WMMA_BF16(A, B, C) \
    __builtin_amdgcn_wmma_f32_16x16x32_bf16(false, (A), false, (B), (short)0, (C), false, false)

// ---------------------------------------------------------------- kernels
__global__ void k_zero(float* acc) {
    if (threadIdx.x < 8) acc[threadIdx.x] = 0.f;
}

// x: NCHW f32 -> xf [N=16384, D=256] row-major, bf16 hi/lo split
__global__ __launch_bounds__(256) void k_prep_x(const float* __restrict__ x,
                                                bf16* __restrict__ xhi,
                                                bf16* __restrict__ xlo) {
    int tid = blockIdx.x * 256 + threadIdx.x;       // n*256 + c
    int n = tid >> 8, c = tid & 255;
    int b = n >> 10, hw = n & 1023;
    float v = x[((size_t)(b * DIM + c)) * HW + hw];
    bf16 h = (bf16)v;
    bf16 l = (bf16)(v - (float)h);
    xhi[tid] = h;
    xlo[tid] = l;
}

// emb -> bf16 hi/lo + 0.5*||e||^2 + 1/||e||^2
__global__ __launch_bounds__(256) void k_prep_emb(const float* __restrict__ e,
                                                  bf16* __restrict__ ehi,
                                                  bf16* __restrict__ elo,
                                                  float* __restrict__ hee,
                                                  float* __restrict__ w2) {
    int k = blockIdx.x, c = threadIdx.x;
    float v = e[(size_t)k * DIM + c];
    bf16 h = (bf16)v;
    bf16 l = (bf16)(v - (float)h);
    ehi[(size_t)k * DIM + c] = h;
    elo[(size_t)k * DIM + c] = l;
    float s = block_reduce_sum(v * v);
    if (threadIdx.x == 0) {
        hee[k] = 0.5f * s;
        w2[k]  = 1.0f / s;
    }
}

// Fused distance GEMM + argmin.  argmin_k ||x-e_k||^2 == argmax_k (x.e_k - 0.5||e_k||^2).
// Wave owns 32 rows (2 A row-tiles, hi+lo resident in 256 VGPRs); each iteration
// covers 32 codes (2 B col-tiles): 8 k-steps x 12 WMMA over 4 independent chains.
// Grid is (row-block, k-split): KSPLIT=4 ranges merged afterwards.
__global__ __launch_bounds__(256) void k_argmin(const bf16* __restrict__ xhi,
                                                const bf16* __restrict__ xlo,
                                                const bf16* __restrict__ ehi,
                                                const bf16* __restrict__ elo,
                                                const float* __restrict__ hee,
                                                float* __restrict__ pval,
                                                int* __restrict__ pidx) {
    const int lane = threadIdx.x & 31;
    const int wave = threadIdx.x >> 5;
    const int rb   = blockIdx.x & (ROW_BLOCKS - 1);
    const int ks   = blockIdx.x / ROW_BLOCKS;
    const int r0   = rb * ROWS_PER_WG + wave * 32;  // this wave's 32 rows
    const int m    = lane & 15;                     // A row / B col within tile
    const int h    = lane >> 4;                     // lane half

    // A fragments, contiguous k order matching B:
    // element s -> k = h*16 + s
    v16bf aHi[2][8], aLo[2][8];
    #pragma unroll
    for (int t = 0; t < 2; ++t) {
        #pragma unroll
        for (int j = 0; j < 8; ++j) {
            const bf16* pa = xhi + (size_t)(r0 + t * 16 + m) * DIM + j * 32 + h * 16;
            v8bf p0 = *(const v8bf*)pa;
            v8bf p1 = *(const v8bf*)(pa + 8);
            aHi[t][j] = __builtin_shufflevector(p0, p1, 0,1,2,3,4,5,6,7,8,9,10,11,12,13,14,15);
            const bf16* pl = xlo + (size_t)(r0 + t * 16 + m) * DIM + j * 32 + h * 16;
            v8bf q0 = *(const v8bf*)pl;
            v8bf q1 = *(const v8bf*)(pl + 8);
            aLo[t][j] = __builtin_shufflevector(q0, q1, 0,1,2,3,4,5,6,7,8,9,10,11,12,13,14,15);
        }
    }

    float best[2][8];
    int   bidx[2][8];
    #pragma unroll
    for (int t = 0; t < 2; ++t)
        #pragma unroll
        for (int j = 0; j < 8; ++j) { best[t][j] = -3.4e38f; bidx[t][j] = 0; }

    const int cbase = ks * CODES_PER_SPLIT;
    for (int it = 0; it < CODES_PER_SPLIT / 32; ++it) {
        const int c0 = cbase + it * 32;
        const bf16* pb0h = ehi + (size_t)(c0 + m) * DIM + h * 16;
        const bf16* pb1h = ehi + (size_t)(c0 + 16 + m) * DIM + h * 16;
        const bf16* pb0l = elo + (size_t)(c0 + m) * DIM + h * 16;
        const bf16* pb1l = elo + (size_t)(c0 + 16 + m) * DIM + h * 16;
        if (it + 1 < CODES_PER_SPLIT / 32) {        // temporal prefetch into near caches
            __builtin_prefetch(pb0h + 32 * DIM, 0, 3);
            __builtin_prefetch(pb0l + 32 * DIM, 0, 3);
        }
        v8f a00 = {0.f,0.f,0.f,0.f,0.f,0.f,0.f,0.f};
        v8f a01 = {0.f,0.f,0.f,0.f,0.f,0.f,0.f,0.f};
        v8f a10 = {0.f,0.f,0.f,0.f,0.f,0.f,0.f,0.f};
        v8f a11 = {0.f,0.f,0.f,0.f,0.f,0.f,0.f,0.f};
        #pragma unroll
        for (int j = 0; j < 8; ++j) {
            v16bf b0h = *(const v16bf*)(pb0h + j * 32);
            v16bf b1h = *(const v16bf*)(pb1h + j * 32);
            v16bf b0l = *(const v16bf*)(pb0l + j * 32);
            v16bf b1l = *(const v16bf*)(pb1l + j * 32);
            // 12 WMMAs over 4 independent accumulator chains
            a00 = WMMA_BF16(aHi[0][j], b0h, a00);
            a01 = WMMA_BF16(aHi[0][j], b1h, a01);
            a10 = WMMA_BF16(aHi[1][j], b0h, a10);
            a11 = WMMA_BF16(aHi[1][j], b1h, a11);
            a00 = WMMA_BF16(aLo[0][j], b0h, a00);
            a01 = WMMA_BF16(aLo[0][j], b1h, a01);
            a10 = WMMA_BF16(aLo[1][j], b0h, a10);
            a11 = WMMA_BF16(aLo[1][j], b1h, a11);
            a00 = WMMA_BF16(aHi[0][j], b0l, a00);
            a01 = WMMA_BF16(aHi[0][j], b1l, a01);
            a10 = WMMA_BF16(aHi[1][j], b0l, a10);
            a11 = WMMA_BF16(aHi[1][j], b1l, a11);
        }
        const float he0 = hee[c0 + m];
        const float he1 = hee[c0 + 16 + m];
        const int code0 = c0 + m;
        const int code1 = c0 + 16 + m;
        #pragma unroll
        for (int j = 0; j < 8; ++j) {               // acc[j] = row (j + 8h), col m
            float v00 = a00[j] - he0;
            if (v00 > best[0][j]) { best[0][j] = v00; bidx[0][j] = code0; }
            float v01 = a01[j] - he1;
            if (v01 > best[0][j]) { best[0][j] = v01; bidx[0][j] = code1; }
            float v10 = a10[j] - he0;
            if (v10 > best[1][j]) { best[1][j] = v10; bidx[1][j] = code0; }
            float v11 = a11[j] - he1;
            if (v11 > best[1][j]) { best[1][j] = v11; bidx[1][j] = code1; }
        }
    }

    // argmax across the 16 lanes of each half (xor < 16 stays within a half)
    #pragma unroll
    for (int off = 1; off < 16; off <<= 1) {
        #pragma unroll
        for (int t = 0; t < 2; ++t) {
            #pragma unroll
            for (int j = 0; j < 8; ++j) {
                float ov = __shfl_xor(best[t][j], off, 32);
                int   oi = __shfl_xor(bidx[t][j], off, 32);
                if (ov > best[t][j] || (ov == best[t][j] && oi < bidx[t][j])) {
                    best[t][j] = ov; bidx[t][j] = oi;
                }
            }
        }
    }
    if (m == 0) {
        #pragma unroll
        for (int t = 0; t < 2; ++t) {
            #pragma unroll
            for (int j = 0; j < 8; ++j) {
                int row = r0 + t * 16 + h * 8 + j;
                pval[(size_t)ks * N_TOK + row] = best[t][j];
                pidx[(size_t)ks * N_TOK + row] = bidx[t][j];
            }
        }
    }
}

// merge the KSPLIT candidates per row (strict >, ties -> lower index = first hit)
__global__ __launch_bounds__(256) void k_merge(const float* __restrict__ pval,
                                               const int* __restrict__ pidx,
                                               int* __restrict__ idx) {
    int n = blockIdx.x * 256 + threadIdx.x;
    float bv = pval[n];
    int   bi = pidx[n];
    #pragma unroll
    for (int s = 1; s < KSPLIT; ++s) {
        float v = pval[(size_t)s * N_TOK + n];
        int   i2 = pidx[(size_t)s * N_TOK + n];
        if (v > bv || (v == bv && i2 < bi)) { bv = v; bi = i2; }
    }
    idx[n] = bi;
}

// ||W_n W_n^T - I_K||_F^2 == ||G||_F^2 - K with G = W_n^T W_n (256x256).
// G[p][q] = sum_i e[i][p]*e[i][q] / ||e_i||^2.  LDS-tiled, plain FMA (only ~1 GFLOP).
__global__ __launch_bounds__(256) void k_gram(const float* __restrict__ e,
                                              const float* __restrict__ w2,
                                              float* __restrict__ acc) {
    const int pb = blockIdx.x >> 4, qb = blockIdx.x & 15;
    const int tp = threadIdx.x >> 4, tq = threadIdx.x & 15;
    __shared__ float sp[32][17], sq[32][17], sw[32];
    float g = 0.f;
    for (int i0 = 0; i0 < KCODES; i0 += 32) {
        int r = threadIdx.x >> 4, c = threadIdx.x & 15;
        sp[r][c]      = e[(size_t)(i0 + r) * DIM + pb * 16 + c];
        sp[r + 16][c] = e[(size_t)(i0 + r + 16) * DIM + pb * 16 + c];
        sq[r][c]      = e[(size_t)(i0 + r) * DIM + qb * 16 + c];
        sq[r + 16][c] = e[(size_t)(i0 + r + 16) * DIM + qb * 16 + c];
        if (threadIdx.x < 32) sw[threadIdx.x] = w2[i0 + threadIdx.x];
        __syncthreads();
        #pragma unroll
        for (int r2 = 0; r2 < 32; ++r2) g = fmaf(sw[r2] * sp[r2][tp], sq[r2][tq], g);
        __syncthreads();
    }
    float t = block_reduce_sum(g * g);
    if (threadIdx.x == 0) atomicAdd(&acc[1], t);
}

// x_q = emb[min_d] back to NCHW; accumulate sum((x_q - x)^2)
__global__ __launch_bounds__(256) void k_gather(const float* __restrict__ x,
                                                const float* __restrict__ e,
                                                const int* __restrict__ idx,
                                                float* __restrict__ xq,
                                                float* __restrict__ acc) {
    const int b = blockIdx.x >> 8, c = blockIdx.x & 255;
    float lm = 0.f;
    #pragma unroll
    for (int s = 0; s < 4; ++s) {
        int hw = s * 256 + threadIdx.x;
        int n  = b * HW + hw;
        int k  = idx[n];
        float q  = e[(size_t)k * DIM + c];
        float xv = x[(size_t)blockIdx.x * HW + hw];
        xq[(size_t)blockIdx.x * HW + hw] = q;
        float d = q - xv;
        lm = fmaf(d, d, lm);
    }
    float t = block_reduce_sum(lm);
    if (threadIdx.x == 0) atomicAdd(&acc[0], t);
}

// loss scalar + min_d (as float) into d_out tail
__global__ __launch_bounds__(256) void k_finalize(const int* __restrict__ idx,
                                                  const float* __restrict__ acc,
                                                  float* __restrict__ out) {
    int t = blockIdx.x * 256 + threadIdx.x;
    if (t < N_TOK) out[NUMEL + 1 + t] = (float)idx[t];
    if (t == 0) {
        float mse   = acc[0] * (1.0f / (float)NUMEL);
        float gfro  = acc[1];
        float ortho = sqrtf(fmaxf(gfro - (float)KCODES, 0.f)) /
                      ((float)KCODES * (float)KCODES);
        out[NUMEL] = (1.0f + BETA_C) * mse + ORTHO_C * ortho;
    }
}

// ---------------------------------------------------------------- launcher
extern "C" void kernel_launch(void* const* d_in, const int* in_sizes, int n_in,
                              void* d_out, int out_size, void* d_ws, size_t ws_size,
                              hipStream_t stream) {
    (void)in_sizes; (void)n_in; (void)out_size; (void)ws_size;
    const float* x   = (const float*)d_in[0];
    const float* emb = (const float*)d_in[1];
    float* out = (float*)d_out;

    char* w = (char*)d_ws;
    bf16*  xhi = (bf16*)(w + OFF_XHI);
    bf16*  xlo = (bf16*)(w + OFF_XLO);
    bf16*  ehi = (bf16*)(w + OFF_EHI);
    bf16*  elo = (bf16*)(w + OFF_ELO);
    float* hee = (float*)(w + OFF_HEE);
    float* w2  = (float*)(w + OFF_W2);
    int*   idx = (int*)(w + OFF_IDX);
    float* pval = (float*)(w + OFF_PVAL);
    int*   pidx = (int*)(w + OFF_PIDX);
    float* acc = (float*)(w + OFF_ACC);

    k_zero<<<1, 32, 0, stream>>>(acc);
    k_prep_x<<<NUMEL / 256, 256, 0, stream>>>(x, xhi, xlo);
    k_prep_emb<<<KCODES, 256, 0, stream>>>(emb, ehi, elo, hee, w2);
    k_argmin<<<ROW_BLOCKS * KSPLIT, 256, 0, stream>>>(xhi, xlo, ehi, elo, hee, pval, pidx);
    k_merge<<<N_TOK / 256, 256, 0, stream>>>(pval, pidx, idx);
    k_gram<<<256, 256, 0, stream>>>(emb, w2, acc);
    k_gather<<<BATCH * DIM, 256, 0, stream>>>(x, emb, idx, out, acc);
    k_finalize<<<(N_TOK + 255) / 256, 256, 0, stream>>>(idx, acc, out);
}
